// EdgeDegreeEmbedding_42305427866224
// MI455X (gfx1250) — matrix operand order repaired and checked by reading, not compiled, not run on hardware
//
#include <hip/hip_runtime.h>
#include <hip/hip_bf16.h>

typedef __attribute__((ext_vector_type(16))) _Float16 v16h;
typedef __attribute__((ext_vector_type(8)))  _Float16 v8h;
typedef __attribute__((ext_vector_type(8)))  float    v8f;

#define LMAX 4
#define CCH 64
#define M_ALL 25
#define TM 16            // edges per block
#define NTHREADS 256     // 8 waves (wave32)

// ---------------- weight prep: f32 [K][N] -> f16 [N][K] (transposed) ----------------
__global__ void prep_weights(const float* __restrict__ w0, const float* __restrict__ w1,
                             const float* __restrict__ w2,
                             _Float16* __restrict__ w0t, _Float16* __restrict__ w1t,
                             _Float16* __restrict__ w2t) {
    int t = blockIdx.x * blockDim.x + threadIdx.x;
    int stride = gridDim.x * blockDim.x;
    // w0: [384][128] -> w0t [128][384]
    for (int i = t; i < 128 * 384; i += stride) {
        int n = i / 384, k = i - n * 384;
        w0t[i] = (_Float16)w0[k * 128 + n];
    }
    // w1: [128][128] -> w1t [128][128]
    for (int i = t; i < 128 * 128; i += stride) {
        int n = i >> 7, k = i & 127;
        w1t[i] = (_Float16)w1[k * 128 + n];
    }
    // w2: [128][320] -> w2t [336][128], rows 320..335 zero (uniform-EXEC padding)
    for (int i = t; i < 336 * 128; i += stride) {
        int n = i >> 7, k = i & 127;
        w2t[i] = (_Float16)((n < 320) ? w2[k * 320 + n] : 0.0f);
    }
}

// ---------------- output zeroing (atomic accumulate target) ----------------
__global__ void zero_out(float* __restrict__ p, int n) {
    int i = blockIdx.x * blockDim.x + threadIdx.x;
    int stride = gridDim.x * blockDim.x;
    float4* p4 = (float4*)p;
    int n4 = n >> 2;
    for (; i < n4; i += stride) p4[i] = make_float4(0.f, 0.f, 0.f, 0.f);
}

// ---------------- WMMA fragment loaders ----------------
// A 16x32 f16 (ISA 7.12.2): lanes 0-15: K=0..7 (v0-3), K=16..23 (v4-7);
//                           lanes 16-31: K=8..15, K=24..31. row m = lane&15.
__device__ inline v16h ld_afrag(const _Float16* base, int stride, int k0, int lane) {
    int m = lane & 15;
    int koff = (lane & 16) ? 8 : 0;
    const _Float16* row = base + m * stride + k0;
    v8h x0 = *(const v8h*)(row + koff);
    v8h x1 = *(const v8h*)(row + 16 + koff);
    return __builtin_shufflevector(x0, x1, 0, 1, 2, 3, 4, 5, 6, 7, 8, 9, 10, 11, 12, 13, 14, 15);
}
// B 32x16 f16: lanes 0-15 hold K=0..7,16..23; lanes 16-31 K=8..15,24..31 (matches A);
// column n = n0 + (lane&15). Weights pre-transposed to [N][K] so this is contiguous.
__device__ inline v16h ld_bfrag(const _Float16* wt, int stride, int n0, int k0, int lane) {
    int n = n0 + (lane & 15);
    int koff = (lane & 16) ? 8 : 0;
    const _Float16* p = wt + n * stride + k0 + koff;
    v8h x0 = *(const v8h*)(p);
    v8h x1 = *(const v8h*)(p + 16);
    return __builtin_shufflevector(x0, x1, 0, 1, 2, 3, 4, 5, 6, 7, 8, 9, 10, 11, 12, 13, 14, 15);
}

// ---------------- LayerNorm + SiLU: sX f32[16][128] -> sH f16[16][Hstride] ----------------
__device__ inline void ln_silu(const float* __restrict__ sX, _Float16* __restrict__ sH,
                               int Hstride, const float* __restrict__ g,
                               const float* __restrict__ be, int wv, int lane) {
    #pragma unroll
    for (int rr = 0; rr < 2; ++rr) {
        int r = wv * 2 + rr;
        float x0 = sX[r * 128 + lane];
        float x1 = sX[r * 128 + lane + 32];
        float x2 = sX[r * 128 + lane + 64];
        float x3 = sX[r * 128 + lane + 96];
        float s = x0 + x1 + x2 + x3;
        float q = x0 * x0 + x1 * x1 + x2 * x2 + x3 * x3;
        #pragma unroll
        for (int off = 16; off > 0; off >>= 1) {
            s += __shfl_xor(s, off, 32);
            q += __shfl_xor(q, off, 32);
        }
        float mean = s * (1.0f / 128.0f);
        float var  = q * (1.0f / 128.0f) - mean * mean;
        float rstd = rsqrtf(var + 1e-5f);
        #pragma unroll
        for (int j = 0; j < 4; ++j) {
            int c = lane + 32 * j;
            float xv = (j == 0) ? x0 : (j == 1) ? x1 : (j == 2) ? x2 : x3;
            float v = (xv - mean) * rstd * g[c] + be[c];
            float h = v / (1.0f + __expf(-v));     // SiLU
            sH[r * Hstride + c] = (_Float16)h;
        }
    }
}

// ---------------- main fused kernel: 16 edges per block ----------------
__launch_bounds__(NTHREADS)
__global__ void edge_degree_kernel(
    const int* __restrict__ atomic_numbers, const float* __restrict__ edge_distance,
    const int* __restrict__ edge_index, const float* __restrict__ envw,
    const float* __restrict__ source_emb, const float* __restrict__ target_emb,
    const _Float16* __restrict__ w0t, const float* __restrict__ b0,
    const float* __restrict__ g0, const float* __restrict__ beta0,
    const _Float16* __restrict__ w1t, const float* __restrict__ b1,
    const float* __restrict__ g1, const float* __restrict__ beta1,
    const _Float16* __restrict__ w2t, const float* __restrict__ b2,
    const float* __restrict__ wig, float* __restrict__ out, int E) {

    __shared__ __align__(16) _Float16 sA[TM * 392];   // concat input, stride 392 (16B-aligned rows)
    __shared__ __align__(16) _Float16 sH[TM * 136];   // hidden f16, stride 136
    __shared__ __align__(16) float    sX[TM * 128];   // pre-LN f32
    __shared__ __align__(16) float    sY[TM * 328];   // GEMM3 out f32, stride 328
    __shared__ __align__(16) float    sWig[TM * 128]; // wigner[:, :, :5] flattened 125/row
    __shared__ float sEnv[TM];
    __shared__ int   sNode[TM];
    __shared__ int   sAN0[TM], sAN1[TM];

    const int tid  = threadIdx.x;
    const int lane = tid & 31;
    const int wv   = tid >> 5;
    const int e0   = blockIdx.x * TM;

    // ---- stage per-edge scalars ----
    if (tid < TM) {
        int e = e0 + tid;
        int ec = (e < E) ? e : (E - 1);
        sEnv[tid]  = envw[ec] * 0.2f;                 // fold 1/RESCALE here
        int sn = edge_index[ec];
        int tn = edge_index[E + ec];
        sNode[tid] = tn;
        sAN0[tid] = atomic_numbers[sn];
        sAN1[tid] = atomic_numbers[tn];
    }
    __syncthreads();

    // ---- build A = concat(dist, src_emb, tgt_emb) as f16 ----
    for (int i = tid; i < TM * 384; i += NTHREADS) {
        int r = i / 384, c = i - r * 384;
        int e = e0 + r;
        int ec = (e < E) ? e : (E - 1);
        float v;
        if (c < 128)      v = edge_distance[ec * 128 + c];
        else if (c < 256) v = source_emb[sAN0[r] * 128 + (c - 128)];
        else              v = target_emb[sAN1[r] * 128 + (c - 256)];
        sA[r * 392 + c] = (_Float16)v;
    }
    // ---- preload wigner_inv[e, m, 0..4] ----
    for (int i = tid; i < TM * 125; i += NTHREADS) {
        int r = i / 125, j = i - r * 125;
        int m = j / 5, l = j - m * 5;
        int e = e0 + r;
        int ec = (e < E) ? e : (E - 1);
        sWig[r * 128 + j] = wig[ec * 625 + m * 25 + l];
    }
    __syncthreads();

    // ---- GEMM1: [16x384] @ [384x128] ----
    {
        v8f acc = {};
        #pragma unroll
        for (int k0 = 0; k0 < 384; k0 += 32) {
            v16h a = ld_afrag(sA, 392, k0, lane);
            v16h b = ld_bfrag(w0t, 384, wv * 16, k0, lane);
            acc = __builtin_amdgcn_wmma_f32_16x16x32_f16(false, a, false, b, (short)0, acc,
                                                         false, false);
        }
        int n = wv * 16 + (lane & 15);
        float bias = b0[n];
        int mo = (lane & 16) ? 8 : 0;
        #pragma unroll
        for (int v = 0; v < 8; ++v) sX[(v + mo) * 128 + n] = acc[v] + bias;
    }
    __syncthreads();
    ln_silu(sX, sH, 136, g0, beta0, wv, lane);
    __syncthreads();

    // ---- GEMM2: [16x128] @ [128x128] ----
    {
        v8f acc = {};
        #pragma unroll
        for (int k0 = 0; k0 < 128; k0 += 32) {
            v16h a = ld_afrag(sH, 136, k0, lane);
            v16h b = ld_bfrag(w1t, 128, wv * 16, k0, lane);
            acc = __builtin_amdgcn_wmma_f32_16x16x32_f16(false, a, false, b, (short)0, acc,
                                                         false, false);
        }
        int n = wv * 16 + (lane & 15);
        float bias = b1[n];
        int mo = (lane & 16) ? 8 : 0;
        #pragma unroll
        for (int v = 0; v < 8; ++v) sX[(v + mo) * 128 + n] = acc[v] + bias;
    }
    __syncthreads();
    ln_silu(sX, sH, 136, g1, beta1, wv, lane);
    __syncthreads();

    // ---- GEMM3: [16x128] @ [128x336(pad)] -> keep first 320, *env/RESCALE ----
    {
        v16h afr[4];
        #pragma unroll
        for (int kk = 0; kk < 4; ++kk) afr[kk] = ld_afrag(sH, 136, kk * 32, lane);
        #pragma unroll
        for (int i = 0; i < 3; ++i) {
            int nb = (wv + 8 * i) * 16;           // 0..368 step 16; only tiles with nb < 320 are stored
            v8f acc = {};
            #pragma unroll
            for (int kk = 0; kk < 4; ++kk) {
                v16h b = ld_bfrag(w2t, 128, nb, kk * 32, lane);
                acc = __builtin_amdgcn_wmma_f32_16x16x32_f16(false, afr[kk], false, b, (short)0,
                                                             acc, false, false);
            }
            if (nb < 320) {
                int n = nb + (lane & 15);
                float bias = b2[n];
                int mo = (lane & 16) ? 8 : 0;
                #pragma unroll
                for (int v = 0; v < 8; ++v) {
                    int r = v + mo;
                    sY[r * 328 + n] = (acc[v] + bias) * sEnv[r];
                }
            }
        }
    }
    __syncthreads();

    // ---- wigner contraction [25x5]@[5x64] + atomic scatter to target node ----
    #pragma unroll
    for (int rr = 0; rr < 2; ++rr) {
        int r = wv * 2 + rr;
        if (e0 + r >= E) continue;
        float y[5][2];
        #pragma unroll
        for (int l = 0; l < 5; ++l) {
            y[l][0] = sY[r * 328 + l * 64 + lane];
            y[l][1] = sY[r * 328 + l * 64 + lane + 32];
        }
        float* op = out + (size_t)sNode[r] * (M_ALL * CCH);
        #pragma unroll
        for (int m = 0; m < M_ALL; ++m) {
            const float* wr = &sWig[r * 128 + m * 5];
            float s0 = 0.f, s1 = 0.f;
            #pragma unroll
            for (int l = 0; l < 5; ++l) {
                float w = wr[l];
                s0 += w * y[l][0];
                s1 += w * y[l][1];
            }
            atomicAdd(op + m * CCH + lane, s0);
            atomicAdd(op + m * CCH + lane + 32, s1);
        }
    }
}

extern "C" void kernel_launch(void* const* d_in, const int* in_sizes, int n_in,
                              void* d_out, int out_size, void* d_ws, size_t ws_size,
                              hipStream_t stream) {
    const int*   atomic_numbers = (const int*)d_in[0];
    const float* edge_distance  = (const float*)d_in[1];
    const int*   edge_index     = (const int*)d_in[2];
    const float* envw           = (const float*)d_in[3];
    const float* source_emb     = (const float*)d_in[4];
    const float* target_emb     = (const float*)d_in[5];
    const float* w0    = (const float*)d_in[6];
    const float* b0    = (const float*)d_in[7];
    const float* g0    = (const float*)d_in[8];
    const float* beta0 = (const float*)d_in[9];
    const float* w1    = (const float*)d_in[10];
    const float* b1    = (const float*)d_in[11];
    const float* g1    = (const float*)d_in[12];
    const float* beta1 = (const float*)d_in[13];
    const float* w2    = (const float*)d_in[14];
    const float* b2    = (const float*)d_in[15];
    const float* wig   = (const float*)d_in[16];
    float* out = (float*)d_out;

    const int E = in_sizes[3];  // edge_envelope_weight has E elements

    // workspace layout: f16 transposed weights
    _Float16* w0t = (_Float16*)d_ws;                                   // 128*384*2 = 98304 B
    _Float16* w1t = (_Float16*)((char*)d_ws + 98304);                  // 128*128*2 = 32768 B
    _Float16* w2t = (_Float16*)((char*)d_ws + 98304 + 32768);          // 336*128*2 = 86016 B

    zero_out<<<4096, NTHREADS, 0, stream>>>(out, out_size);
    prep_weights<<<208, NTHREADS, 0, stream>>>(w0, w1, w2, w0t, w1t, w2t);

    int blocks = (E + TM - 1) / TM;
    edge_degree_kernel<<<blocks, NTHREADS, 0, stream>>>(
        atomic_numbers, edge_distance, edge_index, envw, source_emb, target_emb,
        w0t, b0, g0, beta0, w1t, b1, g1, beta1, w2t, b2, wig, out, E);
}
